// GTPCA_Layer_46875273069228
// MI455X (gfx1250) — matrix operand order repaired and checked by reading, not compiled, not run on hardware
//
#include <hip/hip_runtime.h>

// GTPCA fused kernel for MI455X (gfx1250, wave32, WMMA).
// One workgroup (256 threads = 8 waves) per batch image.
//   Phase 1: normalize weight (scalar), stage 128x128 image into LDS as f16,
//            build 16 pre-swizzled 32x16 Toeplitz B-fragments of the weight.
//   Phase 2: each wave computes one 16-row band of the 113x113 score map as
//            8 tiles of 16x16, each tile = 16 accumulating
//            v_wmma_f32_16x16x32_f16 ops (K=32 Toeplitz trick).
//   Phase 3: block-wide argmax(|score|) with first-occurrence tie-break.
//   Phase 4: write output image directly: smax * wn pasted at argmax, zeros
//            elsewhere (scores never hit HBM; total traffic = in + out only).
// v2: capped unrolling at one tile -> VGPRs < 256 (no s_set_vgpr_msb).
// v3: explicit 2-deep A-fragment prefetch double-buffer so DS loads stay in
//     flight across WMMAs (avoid the s_wait_dscnt 0x0 full-drain before every
//     wmma that register recycling caused in v2).

typedef _Float16 v16h __attribute__((ext_vector_type(16)));
typedef _Float16 v8h  __attribute__((ext_vector_type(8)));
typedef _Float16 v4h  __attribute__((ext_vector_type(4)));
typedef float    v8f  __attribute__((ext_vector_type(8)));

constexpr int IMG_H = 128;
constexpr int IMG_W = 128;
constexpr int SH = 113;            // IMG_H - 16 + 1
constexpr int SW = 113;
constexpr float INV_N = 1.0f / 16384.0f;
constexpr int LDS_STRIDE = 152;    // halves/row: 304B, 16B-aligned, bank-friendly
constexpr int LDS_ROWS = 144;      // 128 data rows + 16 zero pad rows

__launch_bounds__(256)
__global__ void gtpca_fused(const float* __restrict__ in,
                            const float* __restrict__ w,
                            float* __restrict__ out)
{
  __shared__ _Float16 img[LDS_ROWS * LDS_STRIDE];            // 43776 B
  // B fragments: [p][half(lo/hi K-block)][lane][8 halves] -> 16 KB
  __shared__ _Float16 bfrag[16 * 2 * 32 * 8];
  __shared__ float wnF[256];                                 // normalized weight, f32
  __shared__ float redA[256];
  __shared__ float redS[256];
  __shared__ int   redI[256];

  const int t    = threadIdx.x;
  const int b    = blockIdx.x;
  const int lane = t & 31;
  const int wid  = t >> 5;         // wave id 0..7

  // ---- weight normalization: invnorm = 1/sqrt(sum(w^2)/n) ----------------
  const float wv = w[t];           // 256 weight elements, one per thread
  redA[t] = wv * wv;
  __syncthreads();
  for (int s = 128; s > 0; s >>= 1) {
    if (t < s) redA[t] += redA[t + s];
    __syncthreads();
  }
  const float invnorm = rsqrtf(redA[0] * INV_N);
  wnF[t] = wv * invnorm;

  // ---- zero padded LDS image (rows/cols >= 128 stay zero) ----------------
  {
    v4h z = {};
    for (int i = t; i < (LDS_ROWS * LDS_STRIDE) / 4; i += 256)
      *(v4h*)&img[i * 4] = z;
  }
  __syncthreads();  // wnF + zeroed img visible to everyone

  // ---- stage image: f32 global -> f16 LDS --------------------------------
  {
    const float4* src = (const float4*)in + (size_t)b * (IMG_H * IMG_W / 4);
    #pragma unroll
    for (int k = 0; k < 16; ++k) {
      const int idx4 = k * 256 + t;           // 0..4095 float4s
      const float4 v = src[idx4];
      const int row = idx4 >> 5;              // 32 float4 per row
      const int col = (idx4 & 31) << 2;
      v4h h = { (_Float16)v.x, (_Float16)v.y, (_Float16)v.z, (_Float16)v.w };
      *(v4h*)&img[row * LDS_STRIDE + col] = h; // 8B-aligned ds_store_b64
    }
  }

  // ---- build Toeplitz B fragments: B_p[k][n] = wn[p][k-n], band 0<=k-n<16 -
  // B layout (16-bit B, 32x16): lanes 0-15 -> col n=lane, K {0-7,16-23};
  //                             lanes 16-31 -> col n=lane-16, K {8-15,24-31}.
  for (int e2 = t; e2 < 512; e2 += 256) {     // (p, lane) pairs
    const int p = e2 >> 5;
    const int l = e2 & 31;
    const int n = l & 15;
    const int koff = (l < 16) ? 0 : 8;
    #pragma unroll
    for (int part = 0; part < 2; ++part) {
      _Float16* dst = &bfrag[(((p * 2) + part) * 32 + l) * 8];
      #pragma unroll
      for (int e = 0; e < 8; ++e) {
        const int q = (koff + part * 16 + e) - n;
        dst[e] = (q >= 0 && q < 16) ? (_Float16)wnF[p * 16 + q] : (_Float16)0.0f;
      }
    }
  }
  __syncthreads();

  // ---- WMMA score tiles + running argmax ---------------------------------
  // A layout (16-bit A, 16x32): lanes 0-15: M=lane, K {0-7,16-23};
  //                             lanes 16-31: M=lane-16, K {8-15,24-31}.
  const int m   = lane & 15;
  const int hi  = lane >> 4;
  const int kbA = hi * 8;
  const int i0  = wid * 16;        // this wave's score-tile row band

  float bestA = -2.0f, bestS = 0.0f;
  int bestI = 0x7fffffff;

  #pragma unroll 1   // keep VGPRs < 256: one tile (16 WMMAs) live at a time
  for (int tj = 0; tj < 8; ++tj) {
    const int j0 = tj * 16;
    v8f c = {};

    // A-fragment 2-deep prefetch pipeline over weight rows p.
    const _Float16* abase = &img[(i0 + m) * LDS_STRIDE + j0 + kbA];
    v8h alo[2], ahi[2];
    alo[0] = *(const v8h*)(abase);
    ahi[0] = *(const v8h*)(abase + 16);
    alo[1] = *(const v8h*)(abase + LDS_STRIDE);
    ahi[1] = *(const v8h*)(abase + LDS_STRIDE + 16);

    #pragma unroll
    for (int p = 0; p < 16; ++p) {
      v16h A = __builtin_shufflevector(alo[p & 1], ahi[p & 1],
                 0,1,2,3,4,5,6,7,8,9,10,11,12,13,14,15);
      if (p + 2 < 16) {           // prefetch row p+2 into the freed buffer
        const _Float16* anext = abase + (p + 2) * LDS_STRIDE;
        alo[p & 1] = *(const v8h*)(anext);
        ahi[p & 1] = *(const v8h*)(anext + 16);
      }
      const _Float16* brow = &bfrag[((p * 2) * 32 + lane) * 8];
      v8h blo = *(const v8h*)(brow);
      v8h bhi = *(const v8h*)(brow + 32 * 8);
      v16h B = __builtin_shufflevector(blo, bhi,
                 0,1,2,3,4,5,6,7,8,9,10,11,12,13,14,15);
      c = __builtin_amdgcn_wmma_f32_16x16x32_f16(
              false, A, false, B, (short)0, c, false, false);
    }

    // C layout: lanes 0-15: N=lane, M=vgpr; lanes 16-31: N=lane-16, M=vgpr+8
    #pragma unroll
    for (int v = 0; v < 8; ++v) {
      const int row = i0 + v + hi * 8;
      const int col = j0 + m;
      const float score = c[v] * INV_N;
      const bool valid = (row < SH) && (col < SW);
      const float a = valid ? fabsf(score) : -1.0f;
      const int idx = row * SW + col;
      if (a > bestA || (a == bestA && idx < bestI)) {
        bestA = a; bestS = score; bestI = idx;
      }
    }
  }

  // ---- block-wide argmax reduction (first-occurrence tie-break) ----------
  redA[t] = bestA; redS[t] = bestS; redI[t] = bestI;
  __syncthreads();
  for (int s = 128; s > 0; s >>= 1) {
    if (t < s) {
      const float a2 = redA[t + s];
      const int   i2 = redI[t + s];
      if (a2 > redA[t] || (a2 == redA[t] && i2 < redI[t])) {
        redA[t] = a2; redS[t] = redS[t + s]; redI[t] = i2;
      }
    }
    __syncthreads();
  }
  const float smax = redS[0];
  const int fidx = redI[0];
  const int rr = fidx / SW;
  const int cc = fidx % SW;

  // ---- write output: smax * wn at (rr,cc), zeros elsewhere ---------------
  float4* dst = (float4*)out + (size_t)b * (IMG_H * IMG_W / 4);
  #pragma unroll
  for (int k = 0; k < 16; ++k) {
    const int idx4 = k * 256 + t;
    const int row = idx4 >> 5;
    const int col = (idx4 & 31) << 2;
    float o[4];
    #pragma unroll
    for (int j = 0; j < 4; ++j) {
      const unsigned dr = (unsigned)(row - rr);
      const unsigned dc = (unsigned)(col + j - cc);
      o[j] = (dr < 16u && dc < 16u) ? smax * wnF[dr * 16 + dc] : 0.0f;
    }
    float4 ov; ov.x = o[0]; ov.y = o[1]; ov.z = o[2]; ov.w = o[3];
    dst[idx4] = ov;
  }
}

extern "C" void kernel_launch(void* const* d_in, const int* in_sizes, int n_in,
                              void* d_out, int out_size, void* d_ws, size_t ws_size,
                              hipStream_t stream) {
  const float* in = (const float*)d_in[0];   // (B, 128, 128) f32
  const float* w  = (const float*)d_in[1];   // (16, 16) f32
  float* out = (float*)d_out;                // (B, 128, 128) f32
  const int batch = in_sizes[0] / (IMG_H * IMG_W);
  gtpca_fused<<<batch, 256, 0, stream>>>(in, w, out);
  (void)d_ws; (void)ws_size; (void)n_in; (void)out_size;
}
